// Neural_Net_12171937316985
// MI455X (gfx1250) — matrix operand order repaired and checked by reading, not compiled, not run on hardware
//
#include <hip/hip_runtime.h>

// ---------------------------------------------------------------------------
// CDNA5 (gfx1250) fused transformer + MLP, wave32 / WMMA f32_16x16x32_f16
// + async global->LDS staging (GLOBAL_LOAD_ASYNC_TO_LDS_B128, ASYNCcnt)
// ---------------------------------------------------------------------------

typedef _Float16 v16h __attribute__((ext_vector_type(16)));
typedef float    v8f  __attribute__((ext_vector_type(8)));

union FragU { uint4 q[2]; v16h h; };

// ---- CDNA5 async copy: memory -> LDS, 16B per lane, tracked by ASYNCcnt ----
__device__ __forceinline__ void async_ld_b128(void* lds_ptr, const void* gptr) {
    // LDS byte address = low 32 bits of the generic (flat) address per the
    // LDS aperture truncation rule (ISA 10.2).
    unsigned int lds_off = (unsigned int)(unsigned long long)lds_ptr;
    asm volatile("global_load_async_to_lds_b128 %0, %1, off"
                 :
                 : "v"(lds_off), "v"(gptr)
                 : "memory");
}
__device__ __forceinline__ void wait_asynccnt0() {
    asm volatile("s_wait_asynccnt 0" ::: "memory");
}

// A fragment (16x32 f16, row-major source, lda in halves, 16B aligned rows).
// ISA 7.12.2: lanes 0-15 -> M=lane, K=[0..7],[16..23]; lanes 16-31 -> M=lane-16, K=[8..15],[24..31]
__device__ __forceinline__ v16h load_fragA(const _Float16* A, int lda, int mt, int kt, int lane) {
    int m  = mt * 16 + (lane & 15);
    int k0 = kt * 32 + ((lane & 16) ? 8 : 0);
    const _Float16* p = A + m * lda + k0;
    FragU f;
    f.q[0] = *(const uint4*)(p);
    f.q[1] = *(const uint4*)(p + 16);
    return f.h;
}

// B fragment (32x16 f16) from transposed weights Bt[N][Kpad]:
// lane l<16 -> column N=l, K=kt*32+[0..7],[16..23] ; lane 16+l -> column N=l, K=kt*32+[8..15],[24..31]
__device__ __forceinline__ v16h load_fragB(const _Float16* Bt, int ldb, int nt, int kt, int lane) {
    int n  = nt * 16 + (lane & 15);
    int k0 = kt * 32 + ((lane & 16) ? 8 : 0);
    const _Float16* p = Bt + n * ldb + k0;
    FragU f;
    f.q[0] = *(const uint4*)(p);
    f.q[1] = *(const uint4*)(p + 16);
    return f.h;
}

__device__ __forceinline__ v8f tile_mm(const _Float16* A, int lda, const _Float16* Bt, int ldb,
                                       int mt, int nt, int ktiles, int lane) {
    v8f acc = {0.f, 0.f, 0.f, 0.f, 0.f, 0.f, 0.f, 0.f};
    for (int kt = 0; kt < ktiles; ++kt) {
        v16h a = load_fragA(A, lda, mt, kt, lane);
        v16h b = load_fragB(Bt, ldb, nt, kt, lane);
        acc = __builtin_amdgcn_wmma_f32_16x16x32_f16(false, a, false, b, (short)0, acc,
                                                     false, false);
    }
    return acc;
}

enum { EP_NONE = 0, EP_RELU = 1, EP_SIN = 2 };

// D layout (§7.12.2): VGPR r, lanes 0-15 -> (M=r, N=lane); lanes 16-31 -> (M=r+8, N=lane-16)
__device__ __forceinline__ void store_tile_f16(const v8f& acc, _Float16* C, int ldc,
                                               int mt, int nt, const float* bias,
                                               int nmax, int lane, int ep) {
    int col = nt * 16 + (lane & 15);
    if (col >= nmax) return;
    float bv = bias ? bias[col] : 0.f;
    int rbase = mt * 16 + ((lane & 16) ? 8 : 0);
#pragma unroll
    for (int r = 0; r < 8; ++r) {
        float v = acc[r] + bv;
        if (ep == EP_RELU) v = fmaxf(v, 0.f);
        else if (ep == EP_SIN) v = __sinf(v);
        C[(rbase + r) * ldc + col] = (_Float16)v;
    }
}

__device__ __forceinline__ void store_tile_f32(const v8f& acc, float* C, int ldc,
                                               int mt, int nt, const float* bias,
                                               int nmax, int lane) {
    int col = nt * 16 + (lane & 15);
    if (col >= nmax) return;
    float bv = bias[col];
    int rbase = mt * 16 + ((lane & 16) ? 8 : 0);
#pragma unroll
    for (int r = 0; r < 8; ++r)
        C[(rbase + r) * ldc + col] = acc[r] + bv;
}

// ---------------------------------------------------------------------------
// Weight pack: f32 [K][N] row-major  ->  f16 transposed [Npad][Kpad], zero pad
// ---------------------------------------------------------------------------
__global__ void pack_wt(const float* __restrict__ W, _Float16* __restrict__ Wt,
                        int K, int N, int Kpad, int Npad) {
    int idx = blockIdx.x * blockDim.x + threadIdx.x;
    if (idx >= Npad * Kpad) return;
    int n = idx / Kpad, k = idx % Kpad;
    float v = (k < K && n < N) ? W[k * N + n] : 0.f;
    Wt[idx] = (_Float16)v;
}

// ---------------------------------------------------------------------------
// Fused encoder: 2 batch elements (64 rows of flattened [B*S, H]) per block.
// ---------------------------------------------------------------------------
__global__ __launch_bounds__(256)
void encoder_kernel(const float* __restrict__ x, const float* __restrict__ y,
                    const float* __restrict__ seq,
                    const float* __restrict__ embW, const float* __restrict__ embB,
                    const _Float16* __restrict__ qkvoT,
                    const _Float16* __restrict__ w1T, const _Float16* __restrict__ w2T,
                    const float* __restrict__ bq, const float* __restrict__ bk,
                    const float* __restrict__ bv, const float* __restrict__ bo,
                    const float* __restrict__ b1, const float* __restrict__ b2,
                    const float* __restrict__ lng, const float* __restrict__ lnb,
                    _Float16* __restrict__ wsX) {
    __shared__ __align__(16) float    sF[64][20];   // master h (f32)
    __shared__ __align__(16) float    sG[64][20];   // gemm f32 output (residual add)
    __shared__ __align__(16) _Float16 sH[64][32];   // f16 padded mirror of h
    __shared__ __align__(16) _Float16 sQ[64][32];
    __shared__ __align__(16) _Float16 sK[64][32];
    __shared__ __align__(16) _Float16 sV[64][32];
    __shared__ __align__(16) _Float16 sT[64][96];   // attn-out (20) / ffn1 (80), Kpad 96
    __shared__ __align__(16) _Float16 sWbuf[4096];  // staged weight tiles

    const int tid = threadIdx.x, lane = tid & 31, wave = tid >> 5;
    const int b0 = blockIdx.x * 2;

    {   // zero sT so K-padding columns are clean f16 zeros
        uint4 z; z.x = z.y = z.z = z.w = 0u;
        for (int u = tid; u < 384; u += 256) ((uint4*)sT)[u] = z;
    }

    // prefetch layer-0 weights into LDS asynchronously (overlaps embedding VALU)
    {
        const uint4* gw = (const uint4*)(qkvoT);
        for (int u = tid; u < 512; u += 256)
            async_ld_b128((void*)(sWbuf + u * 8), (const void*)(gw + u));
    }

    // ---- embedding: h = seq @ embW + embB (tiny K=5, per-row VALU) ----
    if (tid < 64) {
        int r = tid, gb = b0 + (r >> 5), s = r & 31;
        const float* sp = seq + ((size_t)gb * 32 + s) * 5;
        float xin[5];
#pragma unroll
        for (int i = 0; i < 5; ++i) xin[i] = sp[i];
#pragma unroll
        for (int j = 0; j < 20; ++j) {
            float a = embB[j];
#pragma unroll
            for (int i = 0; i < 5; ++i) a += xin[i] * embW[i * 20 + j];
            sF[r][j] = a; sH[r][j] = (_Float16)a;
        }
#pragma unroll
        for (int j = 20; j < 32; ++j) sH[r][j] = (_Float16)0.f;
    }
    wait_asynccnt0();
    __syncthreads();

    for (int L = 0; L < 2; ++L) {
        if (L > 0) {
            // stage this layer's Wq,Wk,Wv,Wo via async copy
            const uint4* gw = (const uint4*)(qkvoT + L * 4096);
            for (int u = tid; u < 512; u += 256)
                async_ld_b128((void*)(sWbuf + u * 8), (const void*)(gw + u));
            wait_asynccnt0();
            __syncthreads();
        }

        // Q,K,V projections: 3 matrices x (4 mtiles x 2 ntiles) = 24 WMMA tile jobs
        for (int job = wave; job < 24; job += 8) {
            int p = job >> 3, mt = (job >> 1) & 3, nt = job & 1;
            v8f acc = tile_mm(&sH[0][0], 32, sWbuf + p * 1024, 32, mt, nt, 1, lane);
            _Float16* dst = (p == 0) ? &sQ[0][0] : (p == 1) ? &sK[0][0] : &sV[0][0];
            const float* bias = ((p == 0) ? bq : (p == 1) ? bk : bv) + L * 20;
            store_tile_f16(acc, dst, 32, mt, nt, bias, 20, lane, EP_NONE);
        }
        __syncthreads();

        // attention over heads axis is row-local: lane-per-row f32 VALU
        if (tid < 64) {
            int r = tid;
            float q[20], k[20], v[20], ao[20];
#pragma unroll
            for (int c = 0; c < 20; ++c) {
                q[c] = (float)sQ[r][c]; k[c] = (float)sK[r][c]; v[c] = (float)sV[r][c];
            }
            const float scale = 0.31622776601683794f;  // 1/sqrt(HD=10)
#pragma unroll
            for (int hh = 0; hh < 2; ++hh) {
                float s0 = 0.f, s1 = 0.f;
#pragma unroll
                for (int d = 0; d < 10; ++d) {
                    s0 += q[hh * 10 + d] * k[d];
                    s1 += q[hh * 10 + d] * k[10 + d];
                }
                s0 *= scale; s1 *= scale;
                float m = fmaxf(s0, s1);
                float e0 = __expf(s0 - m), e1 = __expf(s1 - m);
                float inv = 1.f / (e0 + e1);
                e0 *= inv; e1 *= inv;
#pragma unroll
                for (int d = 0; d < 10; ++d) ao[hh * 10 + d] = e0 * v[d] + e1 * v[10 + d];
            }
#pragma unroll
            for (int c = 0; c < 20; ++c) sT[r][c] = (_Float16)ao[c];
        }
        __syncthreads();

        // output projection: ao @ Wo + bo -> sG (f32)
        for (int job = wave; job < 8; job += 8) {
            int mt = job >> 1, nt = job & 1;
            v8f acc = tile_mm(&sT[0][0], 96, sWbuf + 3 * 1024, 32, mt, nt, 1, lane);
            store_tile_f32(acc, &sG[0][0], 20, mt, nt, bo + L * 20, 20, lane);
        }
        __syncthreads();

        // LN(h + ao) -> sF, sH
        if (tid < 64) {
            int r = tid;
            float v[20]; float mu = 0.f;
#pragma unroll
            for (int c = 0; c < 20; ++c) { v[c] = sF[r][c] + sG[r][c]; mu += v[c]; }
            mu *= 0.05f;
            float var = 0.f;
#pragma unroll
            for (int c = 0; c < 20; ++c) { float d = v[c] - mu; var += d * d; }
            var *= 0.05f;
            float inv = rsqrtf(var + 1e-5f);
            const float* g = lng + L * 20; const float* bb = lnb + L * 20;
#pragma unroll
            for (int c = 0; c < 20; ++c) {
                float o = (v[c] - mu) * inv * g[c] + bb[c];
                sF[r][c] = o; sH[r][c] = (_Float16)o;
            }
        }
        __syncthreads();

        // FFN1 weights (W1t: [80][32]) via async copy, then h @ W1 + b1, relu
        {
            const uint4* g1 = (const uint4*)(w1T + L * 2560);
            for (int u = tid; u < 320; u += 256)
                async_ld_b128((void*)(sWbuf + u * 8), (const void*)(g1 + u));
            wait_asynccnt0();
        }
        __syncthreads();
        for (int job = wave; job < 20; job += 8) {
            int mt = job / 5, nt = job % 5;
            v8f acc = tile_mm(&sH[0][0], 32, sWbuf, 32, mt, nt, 1, lane);
            store_tile_f16(acc, &sT[0][0], 96, mt, nt, b1 + L * 80, 80, lane, EP_RELU);
        }
        __syncthreads();

        // FFN2 weights (W2t: [32][96], K padded 80->96 with zeros), relu @ W2 + b2 -> sG
        {
            const uint4* g2 = (const uint4*)(w2T + L * 3072);
            for (int u = tid; u < 384; u += 256)
                async_ld_b128((void*)(sWbuf + u * 8), (const void*)(g2 + u));
            wait_asynccnt0();
        }
        __syncthreads();
        for (int job = wave; job < 8; job += 8) {
            int mt = job >> 1, nt = job & 1;
            v8f acc = tile_mm(&sT[0][0], 96, sWbuf, 96, mt, nt, 3, lane);
            store_tile_f32(acc, &sG[0][0], 20, mt, nt, b2 + L * 20, 20, lane);
        }
        __syncthreads();

        // LN(h + ff) -> sF, sH
        if (tid < 64) {
            int r = tid;
            float v[20]; float mu = 0.f;
#pragma unroll
            for (int c = 0; c < 20; ++c) { v[c] = sF[r][c] + sG[r][c]; mu += v[c]; }
            mu *= 0.05f;
            float var = 0.f;
#pragma unroll
            for (int c = 0; c < 20; ++c) { float d = v[c] - mu; var += d * d; }
            var *= 0.05f;
            float inv = rsqrtf(var + 1e-5f);
            const float* g = lng + L * 20; const float* bb = lnb + L * 20;
#pragma unroll
            for (int c = 0; c < 20; ++c) {
                float o = (v[c] - mu) * inv * g[c] + bb[c];
                sF[r][c] = o; sH[r][c] = (_Float16)o;
            }
        }
        __syncthreads();
    }

    // mean-pool over S and emit X = [x, y, z] padded to 32 cols (f16 row-major)
    if (tid < 40) {
        int gl = tid / 20, j = tid % 20;
        float s = 0.f;
        for (int ss = 0; ss < 32; ++ss) s += sF[gl * 32 + ss][j];
        wsX[(size_t)(b0 + gl) * 32 + 2 + j] = (_Float16)(s * (1.f / 32.f));
    }
    if (tid < 2) {
        int gb = b0 + tid;
        wsX[(size_t)gb * 32 + 0] = (_Float16)x[gb];
        wsX[(size_t)gb * 32 + 1] = (_Float16)y[gb];
#pragma unroll
        for (int c = 22; c < 32; ++c) wsX[(size_t)gb * 32 + c] = (_Float16)0.f;
    }
}

// ---------------------------------------------------------------------------
// MLP head: 32 rows per block, weights streamed from L2-resident packed f16.
// ---------------------------------------------------------------------------
__global__ __launch_bounds__(256)
void mlp_kernel(const _Float16* __restrict__ Xb,
                const _Float16* __restrict__ m0T, const _Float16* __restrict__ m1T,
                const _Float16* __restrict__ m2T, const _Float16* __restrict__ m3T,
                const float* __restrict__ mb0, const float* __restrict__ mb1,
                const float* __restrict__ mb2, const float* __restrict__ mb3,
                float* __restrict__ out, int Btot) {
    __shared__ __align__(16) _Float16 sA[32][32];
    __shared__ __align__(16) _Float16 s1[32][256];
    __shared__ __align__(16) _Float16 s2[32][256];
    const int tid = threadIdx.x, lane = tid & 31, wave = tid >> 5;
    const int row0 = blockIdx.x * 32;

    // async-stage the X tile into LDS
    {
        const uint4* gx = (const uint4*)(Xb + (size_t)row0 * 32);
        for (int u = tid; u < 128; u += 256)
            async_ld_b128((void*)(&sA[0][0] + u * 8), (const void*)(gx + u));
    }
    __builtin_prefetch(m1T, 0, 1);
    wait_asynccnt0();
    __syncthreads();

    // layer 0: [32,32] @ m0T[256][32], sin
    for (int job = wave; job < 32; job += 8) {
        int mt = job >> 4, nt = job & 15;
        v8f acc = tile_mm(&sA[0][0], 32, m0T, 32, mt, nt, 1, lane);
        store_tile_f16(acc, &s1[0][0], 256, mt, nt, mb0, 256, lane, EP_SIN);
    }
    __builtin_prefetch(m2T, 0, 1);
    __syncthreads();

    // layer 1: [32,256] @ m1T[256][256], sin
    for (int job = wave; job < 32; job += 8) {
        int mt = job >> 4, nt = job & 15;
        v8f acc = tile_mm(&s1[0][0], 256, m1T, 256, mt, nt, 8, lane);
        store_tile_f16(acc, &s2[0][0], 256, mt, nt, mb1, 256, lane, EP_SIN);
    }
    __builtin_prefetch(m3T, 0, 1);
    __syncthreads();

    // layer 2: [32,256] @ m2T[256][256], sin
    for (int job = wave; job < 32; job += 8) {
        int mt = job >> 4, nt = job & 15;
        v8f acc = tile_mm(&s2[0][0], 256, m2T, 256, mt, nt, 8, lane);
        store_tile_f16(acc, &s1[0][0], 256, mt, nt, mb2, 256, lane, EP_SIN);
    }
    __syncthreads();

    // layer 3: [32,256] @ m3T[16][256] -> [32,3], column-major tuple output
    for (int job = wave; job < 2; job += 8) {
        int mt = job;
        v8f acc = tile_mm(&s1[0][0], 256, m3T, 256, mt, 0, 8, lane);
        int col = lane & 15;
        if (col < 3) {
            float bvv = mb3[col];
            int rb = row0 + mt * 16 + ((lane & 16) ? 8 : 0);
#pragma unroll
            for (int r = 0; r < 8; ++r)
                out[(size_t)col * Btot + rb + r] = acc[r] + bvv;
        }
    }
}

// ---------------------------------------------------------------------------
// Host launcher
// ---------------------------------------------------------------------------
extern "C" void kernel_launch(void* const* d_in, const int* in_sizes, int n_in,
                              void* d_out, int out_size, void* d_ws, size_t ws_size,
                              hipStream_t stream) {
    const float* x    = (const float*)d_in[0];
    const float* y    = (const float*)d_in[1];
    const float* seq  = (const float*)d_in[2];
    const float* embW = (const float*)d_in[3];
    const float* embB = (const float*)d_in[4];
    const float* Wq   = (const float*)d_in[5];
    const float* bq   = (const float*)d_in[6];
    const float* Wk   = (const float*)d_in[7];
    const float* bk   = (const float*)d_in[8];
    const float* Wv   = (const float*)d_in[9];
    const float* bv   = (const float*)d_in[10];
    const float* Wo   = (const float*)d_in[11];
    const float* bo   = (const float*)d_in[12];
    const float* W1   = (const float*)d_in[13];
    const float* b1   = (const float*)d_in[14];
    const float* W2   = (const float*)d_in[15];
    const float* b2   = (const float*)d_in[16];
    const float* lng  = (const float*)d_in[17];
    const float* lnb  = (const float*)d_in[18];
    const float* mW0  = (const float*)d_in[19];
    const float* mb0  = (const float*)d_in[20];
    const float* mW1  = (const float*)d_in[21];
    const float* mb1  = (const float*)d_in[22];
    const float* mW2  = (const float*)d_in[23];
    const float* mb2  = (const float*)d_in[24];
    const float* mW3  = (const float*)d_in[25];
    const float* mb3  = (const float*)d_in[26];

    const int Btot = in_sizes[0];  // 65536

    // workspace layout (f16, 256B-aligned regions)
    char* ws = (char*)d_ws;
    size_t off = 0;
    _Float16* wsX   = (_Float16*)(ws + off); off += (size_t)Btot * 32 * 2;   // padded [B][32]
    _Float16* qkvoT = (_Float16*)(ws + off); off += 2 * 4 * 1024 * 2;        // 2 layers x 4 x [32][32]
    _Float16* w1T   = (_Float16*)(ws + off); off += 2 * 2560 * 2;            // 2 x [80][32]
    _Float16* w2T   = (_Float16*)(ws + off); off += 2 * 3072 * 2;            // 2 x [32][96]
    _Float16* m0T   = (_Float16*)(ws + off); off += 256 * 32 * 2;            // [256][32]
    _Float16* m1T   = (_Float16*)(ws + off); off += 256 * 256 * 2;           // [256][256]
    _Float16* m2T   = (_Float16*)(ws + off); off += 256 * 256 * 2;           // [256][256]
    _Float16* m3T   = (_Float16*)(ws + off); off += 16 * 256 * 2;            // [16][256]

    auto pack = [&](const float* W, _Float16* Wt, int K, int N, int Kpad, int Npad) {
        int tot = Kpad * Npad;
        pack_wt<<<(tot + 255) / 256, 256, 0, stream>>>(W, Wt, K, N, Kpad, Npad);
    };
    for (int i = 0; i < 2; ++i) {
        pack(Wq + i * 400,  qkvoT + i * 4096 + 0,    20, 20, 32, 32);
        pack(Wk + i * 400,  qkvoT + i * 4096 + 1024, 20, 20, 32, 32);
        pack(Wv + i * 400,  qkvoT + i * 4096 + 2048, 20, 20, 32, 32);
        pack(Wo + i * 400,  qkvoT + i * 4096 + 3072, 20, 20, 32, 32);
        pack(W1 + i * 1600, w1T + i * 2560, 20, 80, 32, 80);
        pack(W2 + i * 1600, w2T + i * 3072, 80, 20, 96, 32);
    }
    pack(mW0, m0T, 22, 256, 32, 256);
    pack(mW1, m1T, 256, 256, 256, 256);
    pack(mW2, m2T, 256, 256, 256, 256);
    pack(mW3, m3T, 256, 3, 256, 16);

    encoder_kernel<<<Btot / 2, 256, 0, stream>>>(
        x, y, seq, embW, embB, qkvoT, w1T, w2T,
        bq, bk, bv, bo, b1, b2, lng, lnb, wsX);

    mlp_kernel<<<Btot / 32, 256, 0, stream>>>(
        wsX, m0T, m1T, m2T, m3T, mb0, mb1, mb2, mb3,
        (float*)d_out, Btot);
}
